// WindowAttention_1563368096271
// MI455X (gfx1250) — matrix operand order repaired and checked by reading, not compiled, not run on hardware
//
#include <hip/hip_runtime.h>
#include <hip/hip_bf16.h>

// ---------------- feature probes ----------------
#if __has_builtin(__builtin_amdgcn_global_load_async_to_lds_b128) && \
    __has_builtin(__builtin_amdgcn_s_wait_asynccnt)
#define HAVE_ASYNC_LDS 1
#else
#define HAVE_ASYNC_LDS 0
#endif

// ---------------- types / helpers ----------------
typedef __attribute__((ext_vector_type(16))) __bf16 v16bf;
typedef __attribute__((ext_vector_type(8)))  float  v8f;
typedef int v4i_ __attribute__((vector_size(16)));          // matches builtin's V4i param
typedef __attribute__((address_space(1))) v4i_* gptr_v4i;   // global v4i32*
typedef __attribute__((address_space(3))) v4i_* lptr_v4i;   // LDS v4i32*

struct alignas(16) U128 { unsigned int w[4]; };

#define DEVINL static __device__ __forceinline__

DEVINL unsigned short f2bf(float f) {           // hardware cvt via fptrunc
  union { __bf16 h; unsigned short u; } u;
  u.h = (__bf16)f;
  return u.u;
}
DEVINL unsigned int pack2(float a, float b) {   // packed bf16 pair
  union { __bf16 h[2]; unsigned int u; } u;
  u.h[0] = (__bf16)a; u.h[1] = (__bf16)b;
  return u.u;
}

// A-operand (16-bit A 16x32): lane holds row M = lane%16.
// K elements per lane: [8*half, 8*half+8) then [16+8*half, 16+8*half+8).
DEVINL v16bf load_a(const unsigned short* base, int row, int stride, int kbase, int half) {
  union { v16bf v; U128 q[2]; } r;
  const unsigned short* p = base + row * stride + kbase + 8 * half;
  r.q[0] = *(const U128*)(p);
  r.q[1] = *(const U128*)(p + 16);
  return r.v;
}

// B-operand (16-bit B 32x16) fetched from B^T storage (row = N = lane%16):
// K elements per lane: [8*half, 8*half+8) then [16+8*half, 16+8*half+8).
DEVINL v16bf load_b(const unsigned short* base, int row, int stride, int kbase, int half) {
  union { v16bf v; U128 q[2]; } r;
  const unsigned short* p = base + row * stride + kbase + 8 * half;
  r.q[0] = *(const U128*)(p);
  r.q[1] = *(const U128*)(p + 16);
  return r.v;
}

DEVINL v8f wmma_bf16(v16bf a, v16bf b, v8f c) {
  return __builtin_amdgcn_wmma_f32_16x16x32_bf16(false, a, false, b, (short)0, c, false, false);
}

// ---------------- prep kernels (weights/bias/pos shared by all windows) ----------------
__global__ void convert_bf16(const float* __restrict__ src,
                             unsigned short* __restrict__ dst, int n) {
  int i = blockIdx.x * blockDim.x + threadIdx.x;
  if (i < n) dst[i] = f2bf(src[i]);
}

__global__ void build_bias(const float* __restrict__ rpb, float* __restrict__ biasW) {
  int i = blockIdx.x * blockDim.x + threadIdx.x;      // 8*128*128
  if (i >= 8 * 128 * 128) return;
  int n = i & 127, m = (i >> 7) & 127, h = i >> 14;
  int di = m >> 6, hi = (m >> 3) & 7, wi = m & 7;
  int dj = n >> 6, hj = (n >> 3) & 7, wj = n & 7;
  int idx = (di - dj + 1) * 225 + (hi - hj + 7) * 15 + (wi - wj + 7);
  biasW[i] = rpb[idx * 8 + h];                        // bias[h][query m][key n]
}

__global__ void build_pos(float* __restrict__ posW) {
  int i = blockIdx.x * blockDim.x + threadIdx.x;      // 128*256
  if (i >= 128 * 256) return;
  int c = i & 255, t = i >> 8;
  int s = t & 63;                                     // tiled (1,2,1)
  int yi = s >> 3, xj = s & 7;
  float coord = (c < 128) ? (float)(yi + 1) : (float)(xj + 1);
  int k = (c < 128) ? c : (c - 128);
  float v = coord * (6.283185307179586f / (8.0f + 1e-6f));
  float dimt = powf(10000.0f, (float)(k & ~1) * (1.0f / 128.0f));
  float arg = v / dimt;
  posW[i] = (k & 1) ? cosf(arg) : sinf(arg);
}

// ---------------- X staging: global fp32 (+pos) -> LDS bf16, async when available ----------
template <bool ADDPOS>
DEVINL void stage_x(const float* __restrict__ g, const float* __restrict__ pos,
                    unsigned short* __restrict__ sA, float* __restrict__ sStage, int tid) {
#if HAVE_ASYNC_LDS
  for (int c = 0; c < 2; ++c) {                      // two 64KB fp32 chunks via ASYNCcnt
    const float* gc = g + c * 16384;
    for (int i = tid; i < 4096; i += 256) {
      __builtin_amdgcn_global_load_async_to_lds_b128(
          (gptr_v4i)(gc + i * 4), (lptr_v4i)(sStage + i * 4), 0, 0);
    }
    __builtin_amdgcn_s_wait_asynccnt(0);
    __syncthreads();                                  // all waves' async data visible
    const float4* st = (const float4*)sStage;
    const float4* pp = (const float4*)(pos + c * 16384);
    unsigned int* dst = (unsigned int*)sA + c * 8192;
    for (int i = tid; i < 4096; i += 256) {
      float4 v = st[i];
      if (ADDPOS) { float4 p = pp[i]; v.x += p.x; v.y += p.y; v.z += p.z; v.w += p.w; }
      dst[i * 2 + 0] = pack2(v.x, v.y);
      dst[i * 2 + 1] = pack2(v.z, v.w);
    }
    __syncthreads();                                  // staging reusable next chunk
  }
#else
  const float4* gv = (const float4*)g;
  const float4* pv = (const float4*)pos;
  unsigned int* dst = (unsigned int*)sA;
  for (int i = tid; i < 8192; i += 256) {
    float4 v = gv[i];
    if (ADDPOS) { float4 p = pv[i]; v.x += p.x; v.y += p.y; v.z += p.z; v.w += p.w; }
    dst[i * 2 + 0] = pack2(v.x, v.y);
    dst[i * 2 + 1] = pack2(v.z, v.w);
  }
  __syncthreads();
  (void)sStage;
#endif
}

// ---------------- per-head QKV GEMM: X(128x256) @ W_slice^T -> Q*scale, K, V^T in LDS -----
DEVINL void qkv_head_gemm(const unsigned short* __restrict__ sA,
                          const unsigned short* __restrict__ W,   // 768x256 bf16, row-major
                          unsigned short* sQ, unsigned short* sK, unsigned short* sVt,
                          int h, int wave, int half, int l16) {
  const float qscale = 0.1767766952966369f;   // 32^-0.5 folded into Q (as reference does)
  for (int t = 0; t < 6; ++t) {               // 48 tiles over 8 waves
    const int tile = wave * 6 + t;
    const int sect = tile >> 4;               // 0=Q 1=K 2=V
    const int rem  = tile & 15;
    const int Mb   = (rem >> 1) * 16;
    const int nt   = rem & 1;
    const int wrow = sect * 256 + h * 32 + nt * 16 + l16;
    v8f c = {};
    #pragma unroll
    for (int kk = 0; kk < 8; ++kk) {
      v16bf a = load_a(sA, Mb + l16, 256, kk * 32, half);
      v16bf b = load_b(W,  wrow,     256, kk * 32, half);
      c = wmma_bf16(a, b, c);
    }
    if (sect == 0) {
      #pragma unroll
      for (int r = 0; r < 8; ++r)
        sQ[(Mb + r + 8 * half) * 32 + nt * 16 + l16] = f2bf(c[r] * qscale);
    } else if (sect == 1) {
      #pragma unroll
      for (int r = 0; r < 8; ++r)
        sK[(Mb + r + 8 * half) * 32 + nt * 16 + l16] = f2bf(c[r]);
    } else {                                  // V stored transposed [32][128], packed pairs
      unsigned int* dst = (unsigned int*)(sVt + (nt * 16 + l16) * 128 + Mb + 8 * half);
      #pragma unroll
      for (int rr = 0; rr < 4; ++rr)
        dst[rr] = pack2(c[2 * rr], c[2 * rr + 1]);
    }
  }
}

// ---------------- fused window-attention kernel: 1 block = 1 window ----------------
extern "C" __global__ __launch_bounds__(256, 1)
void fused_window_attention(const float* __restrict__ x,
                            const unsigned short* __restrict__ Wself,
                            const unsigned short* __restrict__ Wmut,
                            const unsigned short* __restrict__ Wp,     // 256x512 bf16
                            const float* __restrict__ biasW,
                            const float* __restrict__ posW,
                            const float* __restrict__ bproj,
                            float* __restrict__ out) {
  extern __shared__ unsigned short lds[];
  unsigned short* sA  = lds;               // 128x256 bf16 (X or X+pos)   64KB
  unsigned short* sQ  = sA + 32768;        // 128x32 (pre-scaled)          8KB
  unsigned short* sK  = sQ + 4096;         // 128x32                       8KB
  unsigned short* sVt = sK + 4096;         // 32x128 (V^T)                 8KB
  unsigned short* sP  = sVt + 4096;        // 128x128 probs               32KB
  unsigned short* sOm = sP + 16384;        // 128x256 mutual out          64KB (early: fp32 staging)
  unsigned short* sOs = sOm + 32768;       // 128x256 self out            64KB

  const int tid  = threadIdx.x;
  const int wave = tid >> 5;
  const int lane = tid & 31;
  const int half = lane >> 4;
  const int l16  = lane & 15;
  const int Mb   = wave * 16;              // this wave's 16 output rows
  const float* __restrict__ xb = x + (size_t)blockIdx.x * (128 * 256);

  // warm L2 for the weight streams (global_prefetch_b8)
  __builtin_prefetch((const char*)Wself + tid * 1536, 0, 1);
  __builtin_prefetch((const char*)Wmut  + tid * 1536, 0, 1);
  __builtin_prefetch((const char*)Wp    + tid * 1024, 0, 1);

  stage_x<false>(xb, posW, sA, (float*)sOm, tid);

  // ================= self attention (bias, full 128 keys) =================
  for (int h = 0; h < 8; ++h) {
    qkv_head_gemm(sA, Wself, sQ, sK, sVt, h, wave, half, l16);
    __syncthreads();

    v8f s[8];
    {
      v16bf a = load_a(sQ, Mb + l16, 32, 0, half);
      #pragma unroll
      for (int nt = 0; nt < 8; ++nt) {
        v16bf b = load_b(sK, nt * 16 + l16, 32, 0, half);
        v8f c = {};
        s[nt] = wmma_bf16(a, b, c);
      }
    }
    const float* __restrict__ bh = biasW + h * (128 * 128);
    float rinv[8];
    #pragma unroll
    for (int r = 0; r < 8; ++r) {
      const int Mrow = Mb + r + 8 * half;
      float m = -1e30f;
      #pragma unroll
      for (int nt = 0; nt < 8; ++nt) {
        float v = s[nt][r] + bh[Mrow * 128 + nt * 16 + l16];  // scale folded into Q
        s[nt][r] = v;
        m = fmaxf(m, v);
      }
      #pragma unroll
      for (int off = 1; off < 16; off <<= 1) m = fmaxf(m, __shfl_xor(m, off, 32));
      float sum = 0.f;
      #pragma unroll
      for (int nt = 0; nt < 8; ++nt) {
        float p = __expf(s[nt][r] - m);
        sum += p;
        sP[Mrow * 128 + nt * 16 + l16] = f2bf(p);
      }
      #pragma unroll
      for (int off = 1; off < 16; off <<= 1) sum += __shfl_xor(sum, off, 32);
      rinv[r] = 1.0f / sum;
    }
    #pragma unroll
    for (int nt2 = 0; nt2 < 2; ++nt2) {     // O = P @ V, normalized by row-sum reciprocal
      v8f c = {};
      #pragma unroll
      for (int kk = 0; kk < 4; ++kk) {
        v16bf a = load_a(sP, Mb + l16, 128, kk * 32, half);
        v16bf b = load_b(sVt, nt2 * 16 + l16, 128, kk * 32, half);
        c = wmma_bf16(a, b, c);
      }
      #pragma unroll
      for (int r = 0; r < 8; ++r)
        sOs[(Mb + r + 8 * half) * 256 + h * 32 + nt2 * 16 + l16] = f2bf(c[r] * rinv[r]);
    }
    __syncthreads();
  }

  // ================= mutual attention (x+pos, swapped 64-token halves) =================
  stage_x<true>(xb, posW, sA, (float*)sOm, tid);      // sOm still unused here

  const int kvb = (Mb >= 64) ? 64 : 0;      // key/value half for this wave's output rows
  for (int h = 0; h < 8; ++h) {
    qkv_head_gemm(sA, Wmut, sQ, sK, sVt, h, wave, half, l16);
    __syncthreads();

    v8f s[4];
    {
      v16bf a = load_a(sQ, (Mb + l16) ^ 64, 32, 0, half);   // query from opposite half
      #pragma unroll
      for (int nt = 0; nt < 4; ++nt) {
        v16bf b = load_b(sK, kvb + nt * 16 + l16, 32, 0, half);
        v8f c = {};
        s[nt] = wmma_bf16(a, b, c);
      }
    }
    float rinv[8];
    #pragma unroll
    for (int r = 0; r < 8; ++r) {
      const int Mrow = Mb + r + 8 * half;
      float m = -1e30f;
      #pragma unroll
      for (int nt = 0; nt < 4; ++nt) m = fmaxf(m, s[nt][r]); // scale already in Q
      #pragma unroll
      for (int off = 1; off < 16; off <<= 1) m = fmaxf(m, __shfl_xor(m, off, 32));
      float sum = 0.f;
      #pragma unroll
      for (int nt = 0; nt < 4; ++nt) {
        float p = __expf(s[nt][r] - m);
        sum += p;
        sP[Mrow * 128 + nt * 16 + l16] = f2bf(p);
      }
      #pragma unroll
      for (int off = 1; off < 16; off <<= 1) sum += __shfl_xor(sum, off, 32);
      rinv[r] = 1.0f / sum;
    }
    #pragma unroll
    for (int nt2 = 0; nt2 < 2; ++nt2) {
      v8f c = {};
      #pragma unroll
      for (int kk = 0; kk < 2; ++kk) {
        v16bf a = load_a(sP, Mb + l16, 128, kk * 32, half);
        v16bf b = load_b(sVt, nt2 * 16 + l16, 128, kvb + kk * 32, half);
        c = wmma_bf16(a, b, c);
      }
      #pragma unroll
      for (int r = 0; r < 8; ++r)
        sOm[(Mb + r + 8 * half) * 256 + h * 32 + nt2 * 16 + l16] = f2bf(c[r] * rinv[r]);
    }
    __syncthreads();
  }

  // ================= projection: [Omut | Oself](128x512) @ Wp^T + b =================
  float* __restrict__ ob = out + (size_t)blockIdx.x * (128 * 256);
  for (int nt = 0; nt < 16; ++nt) {
    v8f c = {};
    #pragma unroll
    for (int kk = 0; kk < 8; ++kk) {        // k = 0..255 -> mutual half
      v16bf a = load_a(sOm, Mb + l16, 256, kk * 32, half);
      v16bf b = load_b(Wp, nt * 16 + l16, 512, kk * 32, half);
      c = wmma_bf16(a, b, c);
    }
    #pragma unroll
    for (int kk = 0; kk < 8; ++kk) {        // k = 256..511 -> self half
      v16bf a = load_a(sOs, Mb + l16, 256, kk * 32, half);
      v16bf b = load_b(Wp, nt * 16 + l16, 512, 256 + kk * 32, half);
      c = wmma_bf16(a, b, c);
    }
    const int ncol = nt * 16 + l16;
    const float bias = bproj[ncol];
    #pragma unroll
    for (int r = 0; r < 8; ++r)
      ob[(Mb + r + 8 * half) * 256 + ncol] = c[r] + bias;
  }
}

// ---------------- launcher ----------------
extern "C" void kernel_launch(void* const* d_in, const int* in_sizes, int n_in,
                              void* d_out, int out_size, void* d_ws, size_t ws_size,
                              hipStream_t stream) {
  (void)in_sizes; (void)n_in; (void)out_size; (void)ws_size;
  const float* x     = (const float*)d_in[0];
  const float* rpb   = (const float*)d_in[1];
  const float* Wqs   = (const float*)d_in[2];
  const float* Wqm   = (const float*)d_in[3];
  const float* Wpf   = (const float*)d_in[4];
  const float* bproj = (const float*)d_in[5];

  char* ws = (char*)d_ws;
  unsigned short* Wself_bf = (unsigned short*)(ws + 0);        // 768*256*2 = 393216
  unsigned short* Wmut_bf  = (unsigned short*)(ws + 393216);   // 393216
  unsigned short* Wp_bf    = (unsigned short*)(ws + 786432);   // 256*512*2 = 262144
  float*          biasW    = (float*)(ws + 1048576);           // 8*128*128*4 = 524288
  float*          posW     = (float*)(ws + 1572864);           // 128*256*4 = 131072

  convert_bf16<<<768, 256, 0, stream>>>(Wqs, Wself_bf, 196608);
  convert_bf16<<<768, 256, 0, stream>>>(Wqm, Wmut_bf, 196608);
  convert_bf16<<<512, 256, 0, stream>>>(Wpf, Wp_bf, 131072);
  build_bias<<<512, 256, 0, stream>>>(rpb, biasW);
  build_pos<<<128, 256, 0, stream>>>(posW);

  const size_t lds_bytes = (size_t)126976 * 2;                 // 253,952 B of the 320KB WGP LDS
  (void)hipFuncSetAttribute((const void*)fused_window_attention,
                            hipFuncAttributeMaxDynamicSharedMemorySize,
                            (int)lds_bytes);
  fused_window_attention<<<2048, 256, lds_bytes, stream>>>(
      x, Wself_bf, Wmut_bf, Wp_bf, biasW, posW, bproj, (float*)d_out);
}
